// CausalPSTABlock_39891656245440
// MI455X (gfx1250) — matrix operand order repaired and checked by reading, not compiled, not run on hardware
//
#include <hip/hip_runtime.h>
#include <math.h>

// ---------------- shapes ----------------
#define BB   8
#define TT   8
#define NN   196
#define DD   768
#define HH   12
#define HD   64
#define D3   2304      // 3*D
#define MLP  3072
#define MTOK 12544     // B*T*N
#define BT   64        // B*T
#define BTH  768       // B*T*H
#define NPAD 224       // N padded to multiple of 32 for K-loop
#define SCALE 0.125f   // 1/sqrt(64)

#define MT 2           // 16-row tiles per wave
#define NT 4           // 16-col tiles per wave

typedef __bf16 v16bf __attribute__((ext_vector_type(16)));
typedef float  v8f   __attribute__((ext_vector_type(8)));

struct __align__(16) U4 { unsigned x, y, z, w; };
union BFrag { v16bf v; U4 u[2]; };

__device__ __forceinline__ unsigned short f2bf(float f) {
  unsigned u = __float_as_uint(f);
  u += 0x7FFFu + ((u >> 16) & 1u);          // round-to-nearest-even
  return (unsigned short)(u >> 16);
}
__device__ __forceinline__ float bf2f(unsigned short h) {
  return __uint_as_float(((unsigned)h) << 16);
}

// ---------------- fp32 -> bf16 convert (weights) ----------------
__global__ void __launch_bounds__(256) cvt_f32_to_bf16(
    const float* __restrict__ src, unsigned short* __restrict__ dst, int n) {
  int i = blockIdx.x * 256 + threadIdx.x;
  if (i < n) dst[i] = f2bf(src[i]);
}

// ---------------- LayerNorm over D=768, one wave per row ----------------
__global__ void __launch_bounds__(256) ln_to_bf16(
    const float* __restrict__ x, const float* __restrict__ g,
    const float* __restrict__ bta, unsigned short* __restrict__ out, int rows) {
  int row  = blockIdx.x * 8 + (threadIdx.x >> 5);
  int lane = threadIdx.x & 31;
  if (row >= rows) return;
  const float* xr = x + (size_t)row * DD;
  float s = 0.f, sq = 0.f;
  for (int c = lane; c < DD; c += 32) { float v = xr[c]; s += v; sq += v * v; }
  for (int o = 16; o; o >>= 1) { s += __shfl_xor(s, o, 32); sq += __shfl_xor(sq, o, 32); }
  float mean = s * (1.f / DD);
  float var  = sq * (1.f / DD) - mean * mean;
  float rstd = rsqrtf(var + 1e-5f);
  unsigned short* orow = out + (size_t)row * DD;
  for (int c = lane; c < DD; c += 32)
    orow[c] = f2bf((xr[c] - mean) * rstd * g[c] + bta[c]);
}

// ---------------- register-blocked WMMA GEMM ----------------
// C[M,N] = act(A[M,K] * W[N,K]^T + bias) (+resid); A,W bf16; fp32 accum.
// Each wave computes a 32x64 block (2x4 grid of 16x16 WMMA tiles) so every
// A fragment is reused 4x and every B fragment 2x (~22 flop/byte at wave level).
__global__ void __launch_bounds__(256) wmma_gemm_bf16(
    const unsigned short* __restrict__ A, const unsigned short* __restrict__ W,
    const float* __restrict__ bias, const float* __restrict__ resid,
    float* __restrict__ outF, unsigned short* __restrict__ outB,
    int M, int N, int K, int act) {
  const int lane = threadIdx.x & 31;
  const int wave = threadIdx.x >> 5;
  const int tilesN = N >> 6;              // 64-col groups
  const int tilesM = (M + 31) >> 5;       // 32-row groups
  int tile = blockIdx.x * 8 + wave;
  if (tile >= tilesM * tilesN) return;    // wave-uniform
  const int tm = tile / tilesN, tn = tile % tilesN;
  const int hi = lane >> 4, lo = lane & 15;

  const unsigned short* aRow[MT];
  const unsigned short* wRow[NT];
#pragma unroll
  for (int mi = 0; mi < MT; ++mi) {
    int r = tm * 32 + mi * 16 + lo;
    if (r > M - 1) r = M - 1;             // clamp keeps EXEC full for WMMA
    aRow[mi] = A + (size_t)r * K;
  }
#pragma unroll
  for (int ni = 0; ni < NT; ++ni) {
    int c = tn * 64 + ni * 16 + lo;
    wRow[ni] = W + (size_t)c * K;
  }

  v8f acc[MT][NT] = {};
  for (int k0 = 0; k0 < K; k0 += 32) {
    BFrag a[MT], b[NT];
#pragma unroll
    for (int mi = 0; mi < MT; ++mi) {
      a[mi].u[0] = *(const U4*)(aRow[mi] + k0 + hi * 8);
      a[mi].u[1] = *(const U4*)(aRow[mi] + k0 + 16 + hi * 8);
    }
#pragma unroll
    for (int ni = 0; ni < NT; ++ni) {
      b[ni].u[0] = *(const U4*)(wRow[ni] + k0 + hi * 8);
      b[ni].u[1] = *(const U4*)(wRow[ni] + k0 + 16 + hi * 8);
    }
    if (k0 + 32 < K) {
#pragma unroll
      for (int mi = 0; mi < MT; ++mi)
        __builtin_prefetch(aRow[mi] + k0 + 32 + hi * 8, 0, 3);
#pragma unroll
      for (int ni = 0; ni < NT; ++ni)
        __builtin_prefetch(wRow[ni] + k0 + 32 + hi * 16, 0, 3);
    }
#pragma unroll
    for (int mi = 0; mi < MT; ++mi)
#pragma unroll
      for (int ni = 0; ni < NT; ++ni)
        acc[mi][ni] = __builtin_amdgcn_wmma_f32_16x16x32_bf16(
            false, a[mi].v, false, b[ni].v, (short)0, acc[mi][ni], false, false);
  }

#pragma unroll
  for (int mi = 0; mi < MT; ++mi) {
#pragma unroll
    for (int ni = 0; ni < NT; ++ni) {
      const int colO = tn * 64 + ni * 16 + lo;
#pragma unroll
      for (int i = 0; i < 8; ++i) {
        int rowO = tm * 32 + mi * 16 + hi * 8 + i;
        if (rowO >= M) continue;
        float v = acc[mi][ni][i];
        if (bias)  v += bias[colO];
        if (act)   v = 0.5f * v * (1.0f + erff(v * 0.70710678118654752f));
        if (resid) v += resid[(size_t)rowO * N + colO];
        if (outF)  outF[(size_t)rowO * N + colO] = v;
        if (outB)  outB[(size_t)rowO * N + colO] = f2bf(v);
      }
    }
  }
}

// ---------------- temporal attention over T=8 (tiny -> VALU) ----------------
__device__ __forceinline__ size_t tqbase(int b, int t, int n, int h) {
  return ((size_t)((b * TT + t) * NN + n)) * D3 + h * HD;
}
__global__ void __launch_bounds__(256) temporal_attn(
    const unsigned short* __restrict__ qkv, unsigned short* __restrict__ attn_out) {
  int idx = blockIdx.x * 256 + threadIdx.x;
  if (idx >= BB * NN * HH * TT) return;
  int ti = idx & 7;
  int h  = (idx >> 3) % HH;
  int bn = idx / (TT * HH);
  int b = bn / NN, n = bn % NN;
  float q[HD];
  {
    const unsigned short* qp = qkv + tqbase(b, ti, n, h);
#pragma unroll
    for (int d = 0; d < HD; ++d) q[d] = bf2f(qp[d]);
  }
  float sc[TT]; float mx = -1e30f;
#pragma unroll
  for (int j = 0; j < TT; ++j) {
    const unsigned short* kp = qkv + tqbase(b, j, n, h) + DD;
    float dot = 0.f;
#pragma unroll
    for (int d = 0; d < HD; ++d) dot += q[d] * bf2f(kp[d]);
    sc[j] = dot * SCALE;
    mx = fmaxf(mx, sc[j]);
  }
  float sum = 0.f;
#pragma unroll
  for (int j = 0; j < TT; ++j) { sc[j] = __expf(sc[j] - mx); sum += sc[j]; }
  float inv = 1.f / sum;
  float acc[HD];
#pragma unroll
  for (int d = 0; d < HD; ++d) acc[d] = 0.f;
#pragma unroll
  for (int j = 0; j < TT; ++j) {
    const unsigned short* vp = qkv + tqbase(b, j, n, h) + 2 * DD;
    float p = sc[j] * inv;
#pragma unroll
    for (int d = 0; d < HD; ++d) acc[d] += p * bf2f(vp[d]);
  }
  unsigned short* o = attn_out + ((size_t)((b * TT + ti) * NN + n)) * DD + h * HD;
#pragma unroll
  for (int d = 0; d < HD; ++d) o[d] = f2bf(acc[d]);
}

// ---------------- spatial scores: per (bt,h) S[196,196] = q k^T * SCALE + lam*A_causal[b] ----------------
__global__ void __launch_bounds__(256) spatial_scores_wmma(
    const unsigned short* __restrict__ qkv, const float* __restrict__ Ac,
    const float* __restrict__ lam, float* __restrict__ scores) {
  const int batch = blockIdx.y;                 // bt*H + h
  const int bt = batch / HH, h = batch % HH;
  const int b = bt / TT;
  const int lane = threadIdx.x & 31, wave = threadIdx.x >> 5;
  int tile = blockIdx.x * 8 + wave;
  if (tile >= 13 * 13) return;
  const int tm = tile / 13, tn = tile % 13;
  const int hi = lane >> 4, lo = lane & 15;
  int i = tm * 16 + lo; if (i > NN - 1) i = NN - 1;
  int j = tn * 16 + lo; if (j > NN - 1) j = NN - 1;
  const unsigned short* qRow = qkv + ((size_t)(bt * NN + i) * D3 + h * HD);
  const unsigned short* kRow = qkv + ((size_t)(bt * NN + j) * D3 + DD + h * HD);
  v8f acc = {};
  for (int k0 = 0; k0 < HD; k0 += 32) {
    BFrag a, bb;
    a.u[0]  = *(const U4*)(qRow + k0 + hi * 8);
    a.u[1]  = *(const U4*)(qRow + k0 + 16 + hi * 8);
    bb.u[0] = *(const U4*)(kRow + k0 + hi * 8);
    bb.u[1] = *(const U4*)(kRow + k0 + 16 + hi * 8);
    acc = __builtin_amdgcn_wmma_f32_16x16x32_bf16(false, a.v, false, bb.v,
                                                  (short)0, acc, false, false);
  }
  const float l = lam[0];
  float* srow = scores + (size_t)batch * NN * NN;
#pragma unroll
  for (int ii = 0; ii < 8; ++ii) {
    int rowO = tm * 16 + hi * 8 + ii;
    int colO = tn * 16 + lo;
    if (rowO < NN && colO < NN)
      srow[(size_t)rowO * NN + colO] =
          acc[ii] * SCALE + l * Ac[((size_t)b * NN + rowO) * NN + colO];
  }
}

// ---------------- softmax over 196, write bf16 row padded to 224 ----------------
__global__ void __launch_bounds__(256) softmax_rows(
    const float* __restrict__ scores, unsigned short* __restrict__ probs) {
  int row  = blockIdx.x * 8 + (threadIdx.x >> 5);
  int lane = threadIdx.x & 31;
  if (row >= BTH * NN) return;
  const float* s = scores + (size_t)row * NN;
  float mx = -1e30f;
  for (int j = lane; j < NN; j += 32) mx = fmaxf(mx, s[j]);
  for (int o = 16; o; o >>= 1) mx = fmaxf(mx, __shfl_xor(mx, o, 32));
  float sum = 0.f;
  for (int j = lane; j < NN; j += 32) sum += __expf(s[j] - mx);
  for (int o = 16; o; o >>= 1) sum += __shfl_xor(sum, o, 32);
  float inv = 1.f / sum;
  unsigned short* p = probs + (size_t)row * NPAD;
  for (int j = lane; j < NPAD; j += 32)
    p[j] = (j < NN) ? f2bf(__expf(s[j] - mx) * inv) : (unsigned short)0;
}

// ---------------- repack v -> vT[batch][hd][jpad] (zero-padded K) ----------------
__global__ void __launch_bounds__(256) repack_vT(
    const unsigned short* __restrict__ qkv, unsigned short* __restrict__ vT) {
  size_t idx = (size_t)blockIdx.x * 256 + threadIdx.x;
  if (idx >= (size_t)BTH * HD * NPAD) return;
  int jp = (int)(idx % NPAD);
  int hd = (int)((idx / NPAD) % HD);
  int batch = (int)(idx / ((size_t)NPAD * HD));
  int bt = batch / HH, h = batch % HH;
  unsigned short v = 0;
  if (jp < NN)
    v = qkv[(size_t)(bt * NN + jp) * D3 + 2 * DD + h * HD + hd];
  vT[idx] = v;
}

// ---------------- spatial AV: per (bt,h) O[196,64] = P[196,224] * vT[64,224]^T ----------------
// Register-blocked 2x4 like the main GEMM: wave = one 32-row block, cols 0..63.
__global__ void __launch_bounds__(256) spatial_av_wmma(
    const unsigned short* __restrict__ probs, const unsigned short* __restrict__ vT,
    unsigned short* __restrict__ attn_out) {
  const int batch = blockIdx.y;
  const int bt = batch / HH, h = batch % HH;
  const int lane = threadIdx.x & 31, wave = threadIdx.x >> 5;
  if (wave >= 7) return;                       // 7 * 32 = 224 >= 196 rows
  const int tm = wave;
  const int hi = lane >> 4, lo = lane & 15;

  const unsigned short* aRow[MT];
  const unsigned short* wRow[NT];
#pragma unroll
  for (int mi = 0; mi < MT; ++mi) {
    int r = tm * 32 + mi * 16 + lo;
    if (r > NN - 1) r = NN - 1;
    aRow[mi] = probs + (size_t)batch * NN * NPAD + (size_t)r * NPAD;
  }
#pragma unroll
  for (int ni = 0; ni < NT; ++ni)
    wRow[ni] = vT + (size_t)batch * HD * NPAD + (size_t)(ni * 16 + lo) * NPAD;

  v8f acc[MT][NT] = {};
  for (int k0 = 0; k0 < NPAD; k0 += 32) {
    BFrag a[MT], b[NT];
#pragma unroll
    for (int mi = 0; mi < MT; ++mi) {
      a[mi].u[0] = *(const U4*)(aRow[mi] + k0 + hi * 8);
      a[mi].u[1] = *(const U4*)(aRow[mi] + k0 + 16 + hi * 8);
    }
#pragma unroll
    for (int ni = 0; ni < NT; ++ni) {
      b[ni].u[0] = *(const U4*)(wRow[ni] + k0 + hi * 8);
      b[ni].u[1] = *(const U4*)(wRow[ni] + k0 + 16 + hi * 8);
    }
#pragma unroll
    for (int mi = 0; mi < MT; ++mi)
#pragma unroll
      for (int ni = 0; ni < NT; ++ni)
        acc[mi][ni] = __builtin_amdgcn_wmma_f32_16x16x32_bf16(
            false, a[mi].v, false, b[ni].v, (short)0, acc[mi][ni], false, false);
  }
#pragma unroll
  for (int mi = 0; mi < MT; ++mi) {
#pragma unroll
    for (int ni = 0; ni < NT; ++ni) {
      const int colO = ni * 16 + lo;           // 0..63 within head
#pragma unroll
      for (int ii = 0; ii < 8; ++ii) {
        int rowO = tm * 32 + mi * 16 + hi * 8 + ii;
        if (rowO < NN)
          attn_out[((size_t)(bt * NN) + rowO) * DD + h * HD + colO] =
              f2bf(acc[mi][ni][ii]);
      }
    }
  }
}

// ---------------- host-side orchestration ----------------
static inline size_t align256(size_t v) { return (v + 255) & ~(size_t)255; }

extern "C" void kernel_launch(void* const* d_in, const int* in_sizes, int n_in,
                              void* d_out, int out_size, void* d_ws, size_t ws_size,
                              hipStream_t stream) {
  (void)in_sizes; (void)n_in; (void)out_size; (void)ws_size;
  const float* x        = (const float*)d_in[0];
  const float* Ac       = (const float*)d_in[1];
  const float* ln1_g    = (const float*)d_in[2];
  const float* ln1_b    = (const float*)d_in[3];
  const float* t_in_w   = (const float*)d_in[4];
  const float* t_in_b   = (const float*)d_in[5];
  const float* t_out_w  = (const float*)d_in[6];
  const float* t_out_b  = (const float*)d_in[7];
  const float* ln2_g    = (const float*)d_in[8];
  const float* ln2_b    = (const float*)d_in[9];
  const float* s_qkv_w  = (const float*)d_in[10];
  const float* s_proj_w = (const float*)d_in[11];
  const float* s_proj_b = (const float*)d_in[12];
  const float* lam      = (const float*)d_in[13];
  const float* ln3_g    = (const float*)d_in[14];
  const float* ln3_b    = (const float*)d_in[15];
  const float* fc1_w    = (const float*)d_in[16];
  const float* fc1_b    = (const float*)d_in[17];
  const float* fc2_w    = (const float*)d_in[18];
  const float* fc2_b    = (const float*)d_in[19];
  float* out = (float*)d_out;
  char*  ws  = (char*)d_ws;

  size_t off = 0;
  auto take = [&](size_t bytes) { size_t o = off; off = align256(off + bytes); return o; };
  // bf16 weights
  size_t o_tin  = take((size_t)D3 * DD * 2);
  size_t o_tout = take((size_t)DD * DD * 2);
  size_t o_sqkv = take((size_t)D3 * DD * 2);
  size_t o_sprj = take((size_t)DD * DD * 2);
  size_t o_fc1  = take((size_t)MLP * DD * 2);
  size_t o_fc2  = take((size_t)DD * MLP * 2);
  // activations
  size_t o_xn   = take((size_t)MTOK * DD * 2);          // LN output (bf16)
  size_t o_qkv  = take((size_t)MTOK * D3 * 2);          // qkv (bf16), reused
  size_t o_attn = take((size_t)MTOK * DD * 2);          // attn out (bf16), reused
  size_t o_x1   = take((size_t)MTOK * DD * 4);          // after temporal block (fp32)
  size_t o_big  = take((size_t)BTH * NN * NN * 4);      // scores fp32, later MLP hidden bf16
  size_t o_prb  = take((size_t)BTH * NN * NPAD * 2);    // probs bf16 (padded)
  size_t o_vT   = take((size_t)BTH * HD * NPAD * 2);    // v^T bf16 (padded)

  unsigned short* wb_tin  = (unsigned short*)(ws + o_tin);
  unsigned short* wb_tout = (unsigned short*)(ws + o_tout);
  unsigned short* wb_sqkv = (unsigned short*)(ws + o_sqkv);
  unsigned short* wb_sprj = (unsigned short*)(ws + o_sprj);
  unsigned short* wb_fc1  = (unsigned short*)(ws + o_fc1);
  unsigned short* wb_fc2  = (unsigned short*)(ws + o_fc2);
  unsigned short* xn      = (unsigned short*)(ws + o_xn);
  unsigned short* qkv     = (unsigned short*)(ws + o_qkv);
  unsigned short* attn    = (unsigned short*)(ws + o_attn);
  float*          x1      = (float*)(ws + o_x1);
  float*          scores  = (float*)(ws + o_big);
  unsigned short* hbuf    = (unsigned short*)(ws + o_big);
  unsigned short* probs   = (unsigned short*)(ws + o_prb);
  unsigned short* vT      = (unsigned short*)(ws + o_vT);

  auto cvt = [&](const float* s, unsigned short* d, int n) {
    cvt_f32_to_bf16<<<(n + 255) / 256, 256, 0, stream>>>(s, d, n);
  };
  cvt(t_in_w,   wb_tin,  D3 * DD);
  cvt(t_out_w,  wb_tout, DD * DD);
  cvt(s_qkv_w,  wb_sqkv, D3 * DD);
  cvt(s_proj_w, wb_sprj, DD * DD);
  cvt(fc1_w,    wb_fc1,  MLP * DD);
  cvt(fc2_w,    wb_fc2,  DD * MLP);

  auto gemm = [&](const unsigned short* A, const unsigned short* W, const float* bias,
                  const float* resid, float* oF, unsigned short* oB,
                  int M, int N, int K, int act) {
    int tiles = ((M + 31) / 32) * (N / 64);   // 32x64 block per wave
    wmma_gemm_bf16<<<(tiles + 7) / 8, 256, 0, stream>>>(A, W, bias, resid, oF, oB,
                                                        M, N, K, act);
  };

  // ---- Stage 1: temporal MHA ----
  ln_to_bf16<<<MTOK / 8, 256, 0, stream>>>(x, ln1_g, ln1_b, xn, MTOK);
  gemm(xn, wb_tin, t_in_b, nullptr, nullptr, qkv, MTOK, D3, DD, 0);
  temporal_attn<<<(BB * NN * HH * TT) / 256, 256, 0, stream>>>(qkv, attn);
  gemm(attn, wb_tout, t_out_b, x, x1, nullptr, MTOK, DD, DD, 0);

  // ---- Stage 2: causal-bias spatial attention ----
  ln_to_bf16<<<MTOK / 8, 256, 0, stream>>>(x1, ln2_g, ln2_b, xn, MTOK);
  gemm(xn, wb_sqkv, nullptr, nullptr, nullptr, qkv, MTOK, D3, DD, 0);
  {
    dim3 g((13 * 13 + 7) / 8, BTH);
    spatial_scores_wmma<<<g, 256, 0, stream>>>(qkv, Ac, lam, scores);
  }
  softmax_rows<<<(BTH * NN) / 8, 256, 0, stream>>>(scores, probs);
  {
    size_t tot = (size_t)BTH * HD * NPAD;
    repack_vT<<<(unsigned)((tot + 255) / 256), 256, 0, stream>>>(qkv, vT);
  }
  {
    dim3 g(1, BTH);
    spatial_av_wmma<<<g, 256, 0, stream>>>(probs, vT, attn);
  }
  gemm(attn, wb_sprj, s_proj_b, x1, out, nullptr, MTOK, DD, DD, 0);  // x2 -> d_out

  // ---- Stage 3: MLP ----
  ln_to_bf16<<<MTOK / 8, 256, 0, stream>>>(out, ln3_g, ln3_b, xn, MTOK);
  gemm(xn, wb_fc1, fc1_b, nullptr, nullptr, hbuf, MTOK, MLP, DD, 1);   // GELU fused
  gemm(hbuf, wb_fc2, fc2_b, out, out, nullptr, MTOK, DD, MLP, 0);      // + residual
}
